// LSTMDecoder_25786983645770
// MI455X (gfx1250) — matrix operand order repaired and checked by reading, not compiled, not run on hardware
//
#include <hip/hip_runtime.h>
#include <hip/hip_bf16.h>
#include <math.h>

typedef __bf16 bf16_t;
typedef __attribute__((ext_vector_type(16))) __bf16 v16bf;
typedef __attribute__((ext_vector_type(8)))  __bf16 v8bf;
typedef __attribute__((ext_vector_type(8)))  float  v8f;

#define B_DIM 256
#define T_DIM 1024
#define H_DIM 256
#define L_DIM 128
#define V_DIM 13
#define G4H   1024   // 4*H

// ---------------------------------------------------------------------------
// A-fragment loader for v_wmma_f32_16x16x32_bf16.
// CDNA5 16-bit A layout (16x32, MxK): lanes 0-15 hold row M=lane, K = k0+{0..7,16..23};
// lanes 16-31 hold row M=lane-16, K = k0+{8..15,24..31}. Two 16B contiguous loads.
// Caller passes base = rowptr + k0 + ((lane&16)?8:0).
__device__ __forceinline__ v16bf load_a_frag(const bf16_t* base) {
    v8bf lo = *(const v8bf*)(base);
    v8bf hi = *(const v8bf*)(base + 16);
    return __builtin_shufflevector(lo, hi, 0,1,2,3,4,5,6,7,8,9,10,11,12,13,14,15);
}

// B fragments are pre-packed so each lane reads 16 contiguous bf16 (32B):
// pack index = ((frag*32 + lane)*16 + je), frag = ntile*KT + ktile,
// source element: n = ntile*16 + (lane&15), k = ktile*32 + (je&7) + ((je&8)?16:0) + ((lane&16)?8:0),
// B[k][n] = W[n][k]  (W row-major [N][K]).
__global__ void k_pack_B(const float* __restrict__ W, bf16_t* __restrict__ pack,
                         int N, int K) {
    int id = blockIdx.x * blockDim.x + threadIdx.x;
    int total = N * K;
    if (id >= total) return;
    const int KT   = K >> 5;
    int frag  = id >> 9;          // / 512 elems per frag
    int e     = id & 511;
    int lane  = e >> 4;
    int je    = e & 15;
    int ntile = frag / KT;
    int ktile = frag % KT;
    int n = ntile * 16 + (lane & 15);
    int k = ktile * 32 + (je & 7) + ((je & 8) ? 16 : 0) + ((lane & 16) ? 8 : 0);
    pack[id] = (bf16_t)W[(size_t)n * K + k];
}

// combined LSTM bias b_ih + b_hh
__global__ void k_bias(const float* __restrict__ b_ih, const float* __restrict__ b_hh,
                       float* __restrict__ bias) {
    int i = blockIdx.x * blockDim.x + threadIdx.x;
    if (i < G4H) bias[i] = b_ih[i] + b_hh[i];
}

__global__ void k_zero_out(float* __restrict__ out) {
    int i = threadIdx.x;
    if (i < B_DIM) out[i] = 0.f;
}

// h0 = relu(latent @ W_lat.T + b_lat); c0 = 0; latent -> bf16 copy for post GEMM
__global__ void k_init(const float* __restrict__ lat, const float* __restrict__ W_lat,
                       const float* __restrict__ b_lat, bf16_t* __restrict__ h0,
                       float* __restrict__ c, bf16_t* __restrict__ latb) {
    int id = blockIdx.x * blockDim.x + threadIdx.x;     // 65536 = B*H
    int b  = id >> 8;
    int hi = id & 255;
    const float* lr = lat + (size_t)b * L_DIM;
    const float* wr = W_lat + (size_t)hi * L_DIM;
    float s = b_lat[hi];
    #pragma unroll 4
    for (int k = 0; k < L_DIM; ++k) s += lr[k] * wr[k];
    s = fmaxf(s, 0.f);
    h0[id] = (bf16_t)s;
    c[id]  = 0.f;
    if (hi < L_DIM) latb[(size_t)b * L_DIM + hi] = (bf16_t)lr[hi];
}

// ---------------------------------------------------------------------------
// One LSTM time step, fully fused:
//   gates[B,4H] = h_prev @ W_hh.T (WMMA bf16)  + x_t @ W_ih.T (K=13, VALU) + bias
//   c,h update fused per-lane (i/f/g/o accumulators share (m,n) positions).
// Wave -> (tile_m over B, tile_n over H); computes 4 gate tiles (N, N+256, N+512, N+768).
__global__ __launch_bounds__(256)
void k_lstm_step(const bf16_t* __restrict__ h_prev, bf16_t* __restrict__ h_next,
                 float* __restrict__ c, const bf16_t* __restrict__ whh_pack,
                 const float* __restrict__ seq, const float* __restrict__ W_ih,
                 const float* __restrict__ bias4h, bf16_t* __restrict__ hs,
                 int t) {
    const int lane = threadIdx.x & 31;
    const int wave = blockIdx.x * 8 + (threadIdx.x >> 5);   // 0..255
    const int tile_m = wave >> 4;                           // B tile
    const int tile_n = wave & 15;                           // H tile

    v8f acc0 = {}, acc1 = {}, acc2 = {}, acc3 = {};         // i, f, g, o
    const bf16_t* arow = h_prev + (size_t)(tile_m * 16 + (lane & 15)) * H_DIM
                                + ((lane & 16) ? 8 : 0);
    #pragma unroll
    for (int kt = 0; kt < 8; ++kt) {
        v16bf a = load_a_frag(arow + kt * 32);
        const size_t lb = (size_t)lane * 16;
        {
            v16bf bb = *(const v16bf*)(whh_pack + (((size_t)(0*16 + tile_n) * 8 + kt) * 32) * 16 + lb);
            acc0 = __builtin_amdgcn_wmma_f32_16x16x32_bf16(false, a, false, bb, (short)0, acc0, false, false);
        }
        {
            v16bf bb = *(const v16bf*)(whh_pack + (((size_t)(1*16 + tile_n) * 8 + kt) * 32) * 16 + lb);
            acc1 = __builtin_amdgcn_wmma_f32_16x16x32_bf16(false, a, false, bb, (short)0, acc1, false, false);
        }
        {
            v16bf bb = *(const v16bf*)(whh_pack + (((size_t)(2*16 + tile_n) * 8 + kt) * 32) * 16 + lb);
            acc2 = __builtin_amdgcn_wmma_f32_16x16x32_bf16(false, a, false, bb, (short)0, acc2, false, false);
        }
        {
            v16bf bb = *(const v16bf*)(whh_pack + (((size_t)(3*16 + tile_n) * 8 + kt) * 32) * 16 + lb);
            acc3 = __builtin_amdgcn_wmma_f32_16x16x32_bf16(false, a, false, bb, (short)0, acc3, false, false);
        }
    }

    const int n = tile_n * 16 + (lane & 15);                // this lane's H index
    float wih0[V_DIM], wih1[V_DIM], wih2[V_DIM], wih3[V_DIM];
    #pragma unroll
    for (int v = 0; v < V_DIM; ++v) {
        wih0[v] = W_ih[(size_t)(0*H_DIM + n) * V_DIM + v];
        wih1[v] = W_ih[(size_t)(1*H_DIM + n) * V_DIM + v];
        wih2[v] = W_ih[(size_t)(2*H_DIM + n) * V_DIM + v];
        wih3[v] = W_ih[(size_t)(3*H_DIM + n) * V_DIM + v];
    }
    const float bg0 = bias4h[0*H_DIM + n], bg1 = bias4h[1*H_DIM + n];
    const float bg2 = bias4h[2*H_DIM + n], bg3 = bias4h[3*H_DIM + n];

    const int mbase = tile_m * 16 + ((lane & 16) ? 8 : 0);  // C layout: M = j + 8*hi-half
    #pragma unroll
    for (int j = 0; j < 8; ++j) {
        const int b = mbase + j;
        const float* xr = seq + ((size_t)b * T_DIM + t) * V_DIM;
        float gi = acc0[j] + bg0, gf = acc1[j] + bg1;
        float gg = acc2[j] + bg2, go = acc3[j] + bg3;
        #pragma unroll
        for (int v = 0; v < V_DIM; ++v) {
            const float x = xr[v];
            gi += x * wih0[v]; gf += x * wih1[v];
            gg += x * wih2[v]; go += x * wih3[v];
        }
        const float si = 1.f / (1.f + __expf(-gi));
        const float sf = 1.f / (1.f + __expf(-gf));
        const float so = 1.f / (1.f + __expf(-go));
        const float tg = tanhf(gg);
        const size_t ci = (size_t)b * H_DIM + n;
        const float cn = sf * c[ci] + si * tg;
        c[ci] = cn;
        const float h = so * tanhf(cn);
        const bf16_t hb = (bf16_t)h;
        h_next[ci] = hb;
        hs[((size_t)b * T_DIM + t) * H_DIM + n] = hb;
    }
}

// ---------------------------------------------------------------------------
// Post pass: one block per 16 rows of [B*T, .]. All 16 rows share batch index b.
//   Z[16,256] = relu([hs|lat] @ W_nl.T + b_nl)   (WMMA bf16, K=384)
//   logits[16,13] = Z @ W_out.T + b_out ; log-softmax ; masked NLL ; atomicAdd(out[b])
__global__ __launch_bounds__(256)
void k_post(const bf16_t* __restrict__ hs, const bf16_t* __restrict__ latb,
            const bf16_t* __restrict__ wnl_pack, const float* __restrict__ b_nl,
            const float* __restrict__ W_out, const float* __restrict__ b_out,
            const int* __restrict__ labels, const int* __restrict__ lengths,
            float* __restrict__ out) {
    __shared__ float Z[16][H_DIM + 4];
    __shared__ float lg[16][16];
    __shared__ float part[16];

    const int rb   = blockIdx.x;        // 0..16383
    const int bidx = rb >> 6;           // T/16 = 64 row-blocks per batch element
    const int t0   = (rb & 63) * 16;
    const int tid  = threadIdx.x;
    const int lane = tid & 31;
    const int wave = tid >> 5;          // 0..7

    const bf16_t* hrow = hs + ((size_t)rb * 16 + (lane & 15)) * H_DIM;
    const bf16_t* lrow = latb + (size_t)bidx * L_DIM;
    const int aoff = (lane & 16) ? 8 : 0;

    #pragma unroll
    for (int half = 0; half < 2; ++half) {
        const int nt = wave + half * 8;             // N tile 0..15
        v8f acc = {};
        #pragma unroll
        for (int kt = 0; kt < 12; ++kt) {
            const bf16_t* abase = (kt < 8) ? (hrow + kt * 32 + aoff)
                                           : (lrow + (kt - 8) * 32 + aoff);
            v16bf a  = load_a_frag(abase);
            v16bf bb = *(const v16bf*)(wnl_pack + (((size_t)(nt * 12 + kt) * 32 + lane) * 16));
            acc = __builtin_amdgcn_wmma_f32_16x16x32_bf16(false, a, false, bb, (short)0, acc, false, false);
        }
        const int n  = nt * 16 + (lane & 15);
        const float bn = b_nl[n];
        const int m0 = (lane & 16) ? 8 : 0;
        #pragma unroll
        for (int j = 0; j < 8; ++j)
            Z[m0 + j][n] = fmaxf(acc[j] + bn, 0.f);
    }
    __syncthreads();

    if (tid < 16 * V_DIM) {
        const int r = tid / V_DIM, v = tid % V_DIM;
        const float* w = W_out + (size_t)v * H_DIM;
        float s = b_out[v];
        #pragma unroll 4
        for (int h = 0; h < H_DIM; ++h) s += Z[r][h] * w[h];
        lg[r][v] = s;
    }
    __syncthreads();

    if (tid < 16) {
        const int t = t0 + tid;
        float m = lg[tid][0];
        #pragma unroll
        for (int v = 1; v < V_DIM; ++v) m = fmaxf(m, lg[tid][v]);
        float se = 0.f;
        #pragma unroll
        for (int v = 0; v < V_DIM; ++v) se += __expf(lg[tid][v] - m);
        const float lse = m + __logf(se);
        const int lab = labels[(size_t)bidx * T_DIM + t];
        const float nll = lse - lg[tid][lab];
        part[tid] = (t < lengths[bidx]) ? nll : 0.f;
    }
    __syncthreads();

    if (tid == 0) {
        float s = 0.f;
        #pragma unroll
        for (int r = 0; r < 16; ++r) s += part[r];
        atomicAdd(&out[bidx], s);
    }
}

// ---------------------------------------------------------------------------
extern "C" void kernel_launch(void* const* d_in, const int* in_sizes, int n_in,
                              void* d_out, int out_size, void* d_ws, size_t ws_size,
                              hipStream_t stream) {
    (void)in_sizes; (void)n_in; (void)out_size; (void)ws_size;
    const float* seq    = (const float*)d_in[0];
    const float* lat    = (const float*)d_in[1];
    const int*   labels = (const int*)  d_in[2];
    const int*   lens   = (const int*)  d_in[3];
    const float* W_lat  = (const float*)d_in[4];
    const float* b_lat  = (const float*)d_in[5];
    const float* W_ih   = (const float*)d_in[6];
    const float* W_hh   = (const float*)d_in[7];
    const float* b_ih   = (const float*)d_in[8];
    const float* b_hh   = (const float*)d_in[9];
    const float* W_nl   = (const float*)d_in[10];
    const float* b_nl   = (const float*)d_in[11];
    const float* W_out  = (const float*)d_in[12];
    const float* b_out  = (const float*)d_in[13];
    float* out = (float*)d_out;

    // workspace carve (all offsets 256B aligned)
    char* w = (char*)d_ws;
    size_t off = 0;
    auto carve = [&](size_t bytes) {
        char* p = w + off;
        off += (bytes + 255) & ~(size_t)255;
        return p;
    };
    bf16_t* hs       = (bf16_t*)carve((size_t)B_DIM * T_DIM * H_DIM * 2); // 128 MB
    bf16_t* hbuf0    = (bf16_t*)carve((size_t)B_DIM * H_DIM * 2);
    bf16_t* hbuf1    = (bf16_t*)carve((size_t)B_DIM * H_DIM * 2);
    float*  cbuf     = (float*) carve((size_t)B_DIM * H_DIM * 4);
    bf16_t* latb     = (bf16_t*)carve((size_t)B_DIM * L_DIM * 2);
    bf16_t* whh_pack = (bf16_t*)carve((size_t)G4H * H_DIM * 2);
    bf16_t* wnl_pack = (bf16_t*)carve((size_t)H_DIM * (H_DIM + L_DIM) * 2);
    float*  bias4h   = (float*) carve((size_t)G4H * 4);

    // prep
    k_bias    <<<4, 256, 0, stream>>>(b_ih, b_hh, bias4h);
    k_pack_B  <<<(G4H * H_DIM + 255) / 256, 256, 0, stream>>>(W_hh, whh_pack, G4H, H_DIM);
    k_pack_B  <<<(H_DIM * (H_DIM + L_DIM) + 255) / 256, 256, 0, stream>>>(W_nl, wnl_pack, H_DIM, H_DIM + L_DIM);
    k_init    <<<(B_DIM * H_DIM) / 256, 256, 0, stream>>>(lat, W_lat, b_lat, hbuf0, cbuf, latb);
    k_zero_out<<<1, 256, 0, stream>>>(out);

    // serial recurrence: one fused WMMA step-kernel per timestep (graph replay
    // amortizes launch cost; double-buffered h avoids cross-block RAW races)
    bf16_t* hb[2] = { hbuf0, hbuf1 };
    for (int t = 0; t < T_DIM; ++t) {
        k_lstm_step<<<32, 256, 0, stream>>>(hb[t & 1], hb[(t + 1) & 1], cbuf,
                                            whh_pack, seq, W_ih, bias4h, hs, t);
    }

    // fused feats-GEMM + vocab projection + log-softmax + masked NLL reduction
    k_post<<<(B_DIM * T_DIM) / 16, 256, 0, stream>>>(hs, latb, wnl_pack, b_nl,
                                                     W_out, b_out, labels, lens, out);
}
